// PacketMambaModel_87368224735535
// MI455X (gfx1250) — matrix operand order repaired and checked by reading, compiled, not run on hardware
//
#include <hip/hip_runtime.h>
#include <hip/hip_bf16.h>
#include <cstdint>
#include <cstddef>

// ---------------- problem constants ----------------
#define BB 8
#define HH 8
#define NN 2048
#define DD 512
#define M3 24            // H*3 memory tokens
#define NSPLIT 8
#define LN_EPS 1e-5f

typedef __bf16 bf16_t;
typedef bf16_t bf16x16 __attribute__((ext_vector_type(16)));
typedef float  f32x8   __attribute__((ext_vector_type(8)));
typedef int    v4i     __attribute__((vector_size(16)));   // matches builtin proto
typedef __attribute__((address_space(3))) v4i* lds_v4i_ptr;

#if defined(__has_builtin)
#if __has_builtin(__builtin_amdgcn_global_load_async_to_lds_b128)
#define USE_ASYNC_LDS 1
#endif
#endif
#ifndef USE_ASYNC_LDS
#define USE_ASYNC_LDS 0
#endif

__device__ __forceinline__ float wave_sum(float v) {
#pragma unroll
  for (int off = 16; off > 0; off >>= 1) v += __shfl_xor(v, off, 32);
  return v;
}
__device__ __forceinline__ float wave_max(float v) {
#pragma unroll
  for (int off = 16; off > 0; off >>= 1) v = fmaxf(v, __shfl_xor(v, off, 32));
  return v;
}

// ---------------------------------------------------------------------------
// Kernel 1: stream history, layer-norm each row, accumulate 3 masked sums in
// registers (thread t owns column d=t). Grid: (B*H, NSPLIT) x 512 threads.
// ---------------------------------------------------------------------------
__global__ __launch_bounds__(512) void k_hist(
    const float* __restrict__ hist, const float* __restrict__ gamma,
    const float* __restrict__ beta, const unsigned char* __restrict__ nmask,
    const int* __restrict__ roles, float* __restrict__ acc_part,
    float* __restrict__ cnt_part) {
  const int bh = blockIdx.x;            // 0..63
  const int split = blockIdx.y;         // 0..NSPLIT-1
  const int b = bh / HH;
  const int t = threadIdx.x;            // d = t
  const int wave = t >> 5, lane = t & 31;
  __shared__ float red[32];
  __shared__ float bc[2];

  const float g = gamma[t], be = beta[t];
  float accH = 0.f, accM = 0.f, accG = 0.f;
  float cH = 0.f, cM = 0.f, cG = 0.f;

  const int rows = NN / NSPLIT;
  const size_t base = ((size_t)bh * NN + (size_t)split * rows) * DD;
  for (int r = 0; r < rows; ++r) {
    const int n = split * rows + r;
    const float x = hist[base + (size_t)r * DD + t];
    const float s1 = wave_sum(x);
    const float s2 = wave_sum(x * x);
    if (lane == 0) { red[wave] = s1; red[16 + wave] = s2; }
    __syncthreads();
    if (wave == 0) {
      float v1 = (lane < 16) ? red[lane] : 0.f;
      float v2 = (lane < 16) ? red[16 + lane] : 0.f;
      v1 = wave_sum(v1); v2 = wave_sum(v2);
      if (lane == 0) {
        const float mu = v1 * (1.f / DD);
        bc[0] = mu;
        bc[1] = rsqrtf(v2 * (1.f / DD) - mu * mu + LN_EPS);
      }
    }
    __syncthreads();
    const float xh = (x - bc[0]) * bc[1] * g + be;
    const unsigned char mm = nmask[(size_t)b * NN + n];
    const int role = roles[(size_t)b * NN + n];
    if (mm) {
      accG += xh; cG += 1.f;
      if (role == 1)      { accH += xh; cH += 1.f; }
      else if (role == 2) { accM += xh; cM += 1.f; }
    }
  }
  const size_t p = ((size_t)bh * NSPLIT + split) * 3 * DD;
  acc_part[p + 0 * DD + t] = accH;
  acc_part[p + 1 * DD + t] = accM;
  acc_part[p + 2 * DD + t] = accG;
  if (t == 0) {
    const size_t cp = ((size_t)bh * NSPLIT + split) * 3;
    cnt_part[cp + 0] = cH; cnt_part[cp + 1] = cM; cnt_part[cp + 2] = cG;
  }
}

// ---------------------------------------------------------------------------
// Kernel 2: fold split partials -> memory[B,24,D] (+round/type emb) + mask.
// ---------------------------------------------------------------------------
__global__ __launch_bounds__(512) void k_finalize(
    const float* __restrict__ acc_part, const float* __restrict__ cnt_part,
    const float* __restrict__ remb, const float* __restrict__ temb,
    float* __restrict__ memory, float* __restrict__ mmask) {
  const int bh = blockIdx.x;
  const int b = bh / HH, h = bh % HH;
  const int t = threadIdx.x;
#pragma unroll
  for (int type = 0; type < 3; ++type) {
    float a = 0.f, cnt = 0.f;
    for (int s = 0; s < NSPLIT; ++s) {
      a += acc_part[((size_t)bh * NSPLIT + s) * 3 * DD + (size_t)type * DD + t];
      cnt += cnt_part[((size_t)bh * NSPLIT + s) * 3 + type];
    }
    const float summ = a / fmaxf(cnt, 1.f);
    const size_t mrow = (size_t)b * M3 + h * 3 + type;
    memory[mrow * DD + t] = summ + remb[(size_t)h * DD + t] + temb[(size_t)type * DD + t];
    if (t == 0) mmask[mrow] = (cnt > 0.f) ? 1.f : 0.f;
  }
}

// ---------------------------------------------------------------------------
// Kernel 3: key/value = memory @ Wk/Wv + bias  (192 rows; Wk/Wv L2-resident).
// ---------------------------------------------------------------------------
__global__ __launch_bounds__(512) void k_kv(
    const float* __restrict__ memory, const float* __restrict__ Wk,
    const float* __restrict__ bk, const float* __restrict__ Wv,
    const float* __restrict__ bv, float* __restrict__ key,
    float* __restrict__ value) {
  __shared__ float smem[DD];
  const int row = blockIdx.x;           // 0..B*24-1
  const int t = threadIdx.x;            // output d
  smem[t] = memory[(size_t)row * DD + t];
  __syncthreads();
  float ka = bk[t], va = bv[t];
  for (int j = 0; j < DD; ++j) {
    const float m = smem[j];
    ka += m * Wk[(size_t)j * DD + t];
    va += m * Wv[(size_t)j * DD + t];
  }
  key[(size_t)row * DD + t] = ka;
  value[(size_t)row * DD + t] = va;
}

// ---------------------------------------------------------------------------
// Kernel 4: LN(current_state) -> bf16 rows for the Wq GEMM.
// ---------------------------------------------------------------------------
__global__ __launch_bounds__(512) void k_ln_cur(
    const float* __restrict__ cur, const float* __restrict__ gamma,
    const float* __restrict__ beta, bf16_t* __restrict__ xln) {
  const int row = blockIdx.x;           // 0..B*N-1
  const int t = threadIdx.x;
  const int wave = t >> 5, lane = t & 31;
  __shared__ float red[32];
  __shared__ float bc[2];
  const float x = cur[(size_t)row * DD + t];
  const float s1 = wave_sum(x);
  const float s2 = wave_sum(x * x);
  if (lane == 0) { red[wave] = s1; red[16 + wave] = s2; }
  __syncthreads();
  if (wave == 0) {
    float v1 = (lane < 16) ? red[lane] : 0.f;
    float v2 = (lane < 16) ? red[16 + lane] : 0.f;
    v1 = wave_sum(v1); v2 = wave_sum(v2);
    if (lane == 0) {
      const float mu = v1 * (1.f / DD);
      bc[0] = mu;
      bc[1] = rsqrtf(v2 * (1.f / DD) - mu * mu + LN_EPS);
    }
  }
  __syncthreads();
  xln[(size_t)row * DD + t] = (bf16_t)((x - bc[0]) * bc[1] * gamma[t] + beta[t]);
}

// ---------------------------------------------------------------------------
// f32 [K x N] -> bf16 transposed [N x K] (so GEMM B-tiles are row-contiguous
// and can be staged with async b128 copies, no in-LDS transpose needed).
// ---------------------------------------------------------------------------
__global__ __launch_bounds__(256) void k_cvt_t(const float* __restrict__ in,
                                               bf16_t* __restrict__ out,
                                               int Kdim, int Ndim) {
  __shared__ float tile[32][33];
  const int kb = blockIdx.x * 32, nb = blockIdx.y * 32;
  const int tx = threadIdx.x & 31, ty = threadIdx.x >> 5;   // 8 rows
#pragma unroll
  for (int r = 0; r < 32; r += 8)
    tile[ty + r][tx] = in[(size_t)(kb + ty + r) * Ndim + nb + tx];
  __syncthreads();
#pragma unroll
  for (int r = 0; r < 32; r += 8)
    out[(size_t)(nb + ty + r) * Kdim + kb + tx] = (bf16_t)tile[tx][ty + r];
}

// ---------------------------------------------------------------------------
// WMMA bf16 GEMM:  C[M,N] = A[M,K] @ Bt[N,K]^T + bias, fused epilogues.
// Block 256 thr (8 waves), block tile 128x128, BK=32.
// Wave tile 32x64 -> 8 x v_wmma_f32_16x16x32_bf16 per K-chunk with 4x A-frag
// and 2x B-frag reuse. Tiles staged global->LDS via async b128 DMA
// (ASYNCcnt + barrier), else uint4 copies.
// LDS row stride 40 bf16 = 80B: 16B-aligned rows for B128, even K-pairs stay
// 4B-aligned for ds_load_2addr_b32 fragment reads.
//   EPI 0: Cf = acc + bias                          (query)
//   EPI 1: Cbf = bf16( sigmoid(acc+bias) * aux0 )   (gate * message)
//   EPI 2: Cf = aux0 + acc + bias                   (residual output)
// ---------------------------------------------------------------------------
#define LDT 40
template <int EPI>
__global__ __launch_bounds__(256) void gemm_wmma(
    const bf16_t* __restrict__ A, const bf16_t* __restrict__ Bt,
    const float* __restrict__ bias, int M, int K, int N,
    float* __restrict__ Cf, bf16_t* __restrict__ Cbf,
    const float* __restrict__ aux0) {
  __shared__ bf16_t As[128 * LDT];
  __shared__ bf16_t Bs[128 * LDT];
  const int t = threadIdx.x;
  const int wave = t >> 5, lane = t & 31;
  const int wm = wave & 3;        // 4 m-strips of 32
  const int wn = wave >> 2;       // 2 n-strips of 64
  const int m0 = blockIdx.x * 128, n0 = blockIdx.y * 128;

  f32x8 c[2][4];
#pragma unroll
  for (int s = 0; s < 2; ++s)
#pragma unroll
    for (int u = 0; u < 4; ++u) c[s][u] = (f32x8){};

  for (int k0 = 0; k0 < K; k0 += 32) {
    // stage A and B tiles: 128 rows x 32 bf16 each; 16B chunk per lane-step
#pragma unroll
    for (int r = 0; r < 2; ++r) {
      const int ch = r * 256 + t;        // 0..511
      const int mi = ch >> 2, cc = ch & 3;
      const bf16_t* gA = A + (size_t)(m0 + mi) * K + k0 + cc * 8;
      const bf16_t* gB = Bt + (size_t)(n0 + mi) * K + k0 + cc * 8;
      bf16_t* lA = &As[mi * LDT + cc * 8];
      bf16_t* lB = &Bs[mi * LDT + cc * 8];
#if USE_ASYNC_LDS
      __builtin_amdgcn_global_load_async_to_lds_b128(
          (v4i*)gA, (lds_v4i_ptr)lA, 0, 0);
      __builtin_amdgcn_global_load_async_to_lds_b128(
          (v4i*)gB, (lds_v4i_ptr)lB, 0, 0);
#else
      *(uint4*)lA = *(const uint4*)gA;
      *(uint4*)lB = *(const uint4*)gB;
#endif
    }
#if USE_ASYNC_LDS
    asm volatile("s_wait_asynccnt 0x0" ::: "memory");
#endif
    __syncthreads();

    // A fragments (2): lane&15 = M row; half-lanes K 0-7/16-23 vs 8-15/24-31
    bf16x16 af[2];
    {
      const int r = lane & 15, half = lane >> 4;
#pragma unroll
      for (int s = 0; s < 2; ++s) {
        const int rb = (wm * 32 + s * 16 + r) * LDT;
#pragma unroll
        for (int g = 0; g < 8; ++g) {
          const int kb = (g < 4 ? 2 * g : 16 + 2 * (g - 4)) + 8 * half;
          af[s][2 * g]     = As[rb + kb];
          af[s][2 * g + 1] = As[rb + kb + 1];
        }
      }
    }
    // B fragments (4): lane&15 = N col; half-lanes K 0-7/16-23 vs 8-15/24-31
    bf16x16 bfr[4];
    {
      const int nn = lane & 15, khalf = lane >> 4;
#pragma unroll
      for (int u = 0; u < 4; ++u) {
        const int rb = (wn * 64 + u * 16 + nn) * LDT;
#pragma unroll
        for (int g = 0; g < 8; ++g) {
          const int kk = (g < 4 ? 2 * g : 16 + 2 * (g - 4)) + 8 * khalf;
          bfr[u][2 * g]     = Bs[rb + kk];
          bfr[u][2 * g + 1] = Bs[rb + kk + 1];
        }
      }
    }
#pragma unroll
    for (int s = 0; s < 2; ++s)
#pragma unroll
      for (int u = 0; u < 4; ++u)
        c[s][u] = __builtin_amdgcn_wmma_f32_16x16x32_bf16(
            false, af[s], false, bfr[u], (short)0, c[s][u], false, false);
    __syncthreads();
  }

  // C layout: vgpr j, lane l -> m = j + 8*(l>>4), n = l&15
  const int nloc = lane & 15;
  const int mbase = 8 * (lane >> 4);
#pragma unroll
  for (int s = 0; s < 2; ++s) {
#pragma unroll
    for (int u = 0; u < 4; ++u) {
#pragma unroll
      for (int j = 0; j < 8; ++j) {
        const int gm = m0 + wm * 32 + s * 16 + mbase + j;
        const int gn = n0 + wn * 64 + u * 16 + nloc;
        const float v = c[s][u][j] + bias[gn];
        const size_t idx = (size_t)gm * N + gn;
        if (EPI == 0) {
          Cf[idx] = v;
        } else if (EPI == 1) {
          const float gt = 1.f / (1.f + __expf(-v));
          Cbf[idx] = (bf16_t)(gt * aux0[idx]);
        } else {
          Cf[idx] = aux0[idx] + v;
        }
      }
    }
  }
}

// ---------------------------------------------------------------------------
// Kernel: attention over 24 memory tokens. Block = 256 thr, 32 rows/block.
// key (then value) staged in LDS with pad-513 stride; per-wave softmax via
// shfl over 24 active lanes; deterministic per-block scalar partials.
// ---------------------------------------------------------------------------
__global__ __launch_bounds__(256) void k_attn(
    const float* __restrict__ q, const float* __restrict__ key,
    const float* __restrict__ value, const float* __restrict__ mmask,
    const unsigned char* __restrict__ nmask, const float* __restrict__ cur,
    float* __restrict__ msg, bf16_t* __restrict__ xg,
    float* __restrict__ partials) {
  __shared__ float skv[M3 * 513];
  __shared__ float sm[M3];
  __shared__ float wall[32 * M3];
  __shared__ float wpart[8 * 5];
  const int b = blockIdx.y;
  const int rowblk = blockIdx.x;        // 0..63
  const int t = threadIdx.x, wave = t >> 5, lane = t & 31;

  if (t < M3) sm[t] = mmask[(size_t)b * M3 + t];
  for (int e = t; e < M3 * DD; e += 256) {
    const int m = e >> 9, d = e & 511;
    skv[m * 513 + d] = key[((size_t)b * M3 + m) * DD + d];
  }
  __syncthreads();

  const float scale = 0.044194173824159216f;   // 1/sqrt(512)
  float pent = 0.f, plat = 0.f, phub = 0.f, pmon = 0.f, pglob = 0.f;

#pragma unroll
  for (int rr = 0; rr < 4; ++rr) {
    const int riw = wave * 4 + rr;
    const int i = rowblk * 32 + riw;
    const size_t qoff = ((size_t)b * NN + i) * DD;
    float acc = 0.f;
    if (lane < M3) {
      const float* kr = &skv[lane * 513];
      for (int j = 0; j < DD; ++j) acc += q[qoff + j] * kr[j];
    }
    const bool rowvalid = nmask[(size_t)b * NN + i] != 0;
    const bool pm = rowvalid && (lane < M3) && (sm[lane < M3 ? lane : 0] > 0.f);
    float logit = (lane < M3) ? acc * scale : -3.0e38f;
    if (lane < M3 && !pm) logit = -10000.f;

    const float mx = wave_max(logit);
    const float ex = (lane < M3) ? __expf(logit - mx) : 0.f;
    const float se = wave_sum(ex);
    float w = (lane < M3) ? ex / se : 0.f;
    if (!pm) w = 0.f;

    if (lane < M3) wall[riw * M3 + lane] = w;

    const float ent = w * __logf(w + 1e-8f);
    const float lat = (lane >= 21 && lane < M3) ? w : 0.f;
    const int tp = lane % 3;
    const float hub = (lane < M3 && tp == 0) ? w : 0.f;
    const float mon = (lane < M3 && tp == 1) ? w : 0.f;
    const float glo = (lane < M3 && tp == 2) ? w : 0.f;
    pent += wave_sum(ent);
    plat += wave_sum(lat);
    phub += wave_sum(hub);
    pmon += wave_sum(mon);
    pglob += wave_sum(glo);
  }
  __syncthreads();

  for (int e = t; e < M3 * DD; e += 256) {
    const int m = e >> 9, d = e & 511;
    skv[m * 513 + d] = value[((size_t)b * M3 + m) * DD + d];
  }
  __syncthreads();

#pragma unroll
  for (int rr = 0; rr < 4; ++rr) {
    const int riw = wave * 4 + rr;
    const int i = rowblk * 32 + riw;
    const size_t off = ((size_t)b * NN + i) * DD;
    const size_t goff = ((size_t)b * NN + i) * (2 * DD);
#pragma unroll 4
    for (int kk = 0; kk < 16; ++kk) {
      const int d = lane + kk * 32;
      float a = 0.f;
#pragma unroll
      for (int m = 0; m < M3; ++m) a += wall[riw * M3 + m] * skv[m * 513 + d];
      msg[off + d] = a;
      xg[goff + d] = (bf16_t)cur[off + d];
      xg[goff + DD + d] = (bf16_t)a;
    }
  }

  if (lane == 0) {
    wpart[wave * 5 + 0] = pent; wpart[wave * 5 + 1] = plat;
    wpart[wave * 5 + 2] = phub; wpart[wave * 5 + 3] = pmon;
    wpart[wave * 5 + 4] = pglob;
  }
  __syncthreads();
  if (t == 0) {
    float s[5] = {0.f, 0.f, 0.f, 0.f, 0.f};
    for (int w = 0; w < 8; ++w)
      for (int k = 0; k < 5; ++k) s[k] += wpart[w * 5 + k];
    const int bid = b * 64 + rowblk;
    for (int k = 0; k < 5; ++k) partials[bid * 5 + k] = s[k];
  }
}

// Fixed-order scalar finalize (deterministic).
__global__ void k_scalars(const float* __restrict__ partials,
                          float* __restrict__ out5) {
  if (threadIdx.x == 0 && blockIdx.x == 0) {
    float e = 0.f, l = 0.f, hu = 0.f, mo = 0.f, gl = 0.f;
    for (int i = 0; i < 512; ++i) {
      e += partials[i * 5 + 0]; l += partials[i * 5 + 1];
      hu += partials[i * 5 + 2]; mo += partials[i * 5 + 3];
      gl += partials[i * 5 + 4];
    }
    const float inv = 1.f / (float)(BB * NN);
    out5[0] = -e * inv;  // entropy
    out5[1] = l * inv;   // latest_share
    out5[2] = hu * inv;  // hub_share
    out5[3] = mo * inv;  // monitor_share
    out5[4] = gl * inv;  // global_share
  }
}

// ---------------------------------------------------------------------------
extern "C" void kernel_launch(void* const* d_in, const int* in_sizes, int n_in,
                              void* d_out, int out_size, void* d_ws,
                              size_t ws_size, hipStream_t stream) {
  (void)in_sizes; (void)n_in; (void)out_size; (void)ws_size;
  const float* cur   = (const float*)d_in[0];
  const float* hist  = (const float*)d_in[1];
  const float* lng   = (const float*)d_in[2];
  const float* lnb   = (const float*)d_in[3];
  const float* Wq    = (const float*)d_in[4];
  const float* bq    = (const float*)d_in[5];
  const float* Wk    = (const float*)d_in[6];
  const float* bk    = (const float*)d_in[7];
  const float* Wv    = (const float*)d_in[8];
  const float* bv    = (const float*)d_in[9];
  const float* Wo    = (const float*)d_in[10];
  const float* bo    = (const float*)d_in[11];
  const float* Wg    = (const float*)d_in[12];
  const float* bg    = (const float*)d_in[13];
  const float* remb  = (const float*)d_in[14];
  const float* temb  = (const float*)d_in[15];
  const unsigned char* nmask = (const unsigned char*)d_in[16];  // jax bool_
  const int* roles   = (const int*)d_in[17];
  float* out = (float*)d_out;

  // ---- workspace layout (bytes) ----
  char* ws = (char*)d_ws;
  size_t off = 0;
  auto alloc = [&](size_t bytes) { size_t o = off; off = (off + bytes + 255) & ~(size_t)255; return o; };
  const size_t off_acc  = alloc((size_t)BB * HH * NSPLIT * 3 * DD * 4);
  const size_t off_cnt  = alloc((size_t)BB * HH * NSPLIT * 3 * 4);
  const size_t off_mem  = alloc((size_t)BB * M3 * DD * 4);
  const size_t off_mmk  = alloc((size_t)BB * M3 * 4);
  const size_t off_key  = alloc((size_t)BB * M3 * DD * 4);
  const size_t off_val  = alloc((size_t)BB * M3 * DD * 4);
  const size_t off_wqb  = alloc((size_t)DD * DD * 2);
  const size_t off_wgb  = alloc((size_t)2 * DD * DD * 2);
  const size_t off_wob  = alloc((size_t)DD * DD * 2);
  const size_t off_part = alloc((size_t)512 * 5 * 4);
  const size_t off_xln  = alloc((size_t)BB * NN * DD * 2);
  const size_t off_q    = alloc((size_t)BB * NN * DD * 4);
  const size_t off_msg  = alloc((size_t)BB * NN * DD * 4);
  const size_t off_xg   = alloc((size_t)BB * NN * 2 * DD * 2);
  const size_t off_xu   = alloc((size_t)BB * NN * DD * 2);

  float* acc_part = (float*)(ws + off_acc);
  float* cnt_part = (float*)(ws + off_cnt);
  float* memory   = (float*)(ws + off_mem);
  float* mmask    = (float*)(ws + off_mmk);
  float* key      = (float*)(ws + off_key);
  float* value    = (float*)(ws + off_val);
  bf16_t* wqb     = (bf16_t*)(ws + off_wqb);   // transposed [N x K]
  bf16_t* wgb     = (bf16_t*)(ws + off_wgb);   // transposed [N x 2K]
  bf16_t* wob     = (bf16_t*)(ws + off_wob);   // transposed [N x K]
  float* partials = (float*)(ws + off_part);
  bf16_t* xln     = (bf16_t*)(ws + off_xln);
  float* qbuf     = (float*)(ws + off_q);
  float* msg      = (float*)(ws + off_msg);
  bf16_t* xg      = (bf16_t*)(ws + off_xg);
  bf16_t* xu      = (bf16_t*)(ws + off_xu);

  // 1) history LN + masked partial sums
  k_hist<<<dim3(BB * HH, NSPLIT), 512, 0, stream>>>(hist, lng, lnb, nmask,
                                                    roles, acc_part, cnt_part);
  // 2) memory tokens + mask
  k_finalize<<<BB * HH, 512, 0, stream>>>(acc_part, cnt_part, remb, temb,
                                          memory, mmask);
  // 3) key / value (tiny GEMM, weights L2-resident)
  k_kv<<<BB * M3, 512, 0, stream>>>(memory, Wk, bk, Wv, bv, key, value);
  // 4) LN(current) -> bf16
  k_ln_cur<<<BB * NN, 512, 0, stream>>>(cur, lng, lnb, xln);
  // 5) weight conversions (transposed bf16 for async-friendly B tiles)
  k_cvt_t<<<dim3(DD / 32, DD / 32), 256, 0, stream>>>(Wq, wqb, DD, DD);
  k_cvt_t<<<dim3(2 * DD / 32, DD / 32), 256, 0, stream>>>(Wg, wgb, 2 * DD, DD);
  k_cvt_t<<<dim3(DD / 32, DD / 32), 256, 0, stream>>>(Wo, wob, DD, DD);
  // 6) q = LN(cur) @ Wq + bq   [WMMA]
  gemm_wmma<0><<<dim3((BB * NN) / 128, DD / 128), 256, 0, stream>>>(
      xln, wqb, bq, BB * NN, DD, DD, qbuf, nullptr, nullptr);
  // 7) attention (logits/softmax/message/stats, builds concat bf16)
  k_attn<<<dim3(NN / 32, BB), 256, 0, stream>>>(qbuf, key, value, mmask, nmask,
                                                cur, msg, xg, partials);
  // 8) gated = sigmoid(concat @ Wg + bg) * message   [WMMA, K=1024]
  gemm_wmma<1><<<dim3((BB * NN) / 128, DD / 128), 256, 0, stream>>>(
      xg, wgb, bg, BB * NN, 2 * DD, DD, nullptr, xu, msg);
  // 9) out = cur + gated @ Wo + bo   [WMMA]
  gemm_wmma<2><<<dim3((BB * NN) / 128, DD / 128), 256, 0, stream>>>(
      xu, wob, bo, BB * NN, DD, DD, out, nullptr, cur);
  // 10) scalars
  k_scalars<<<1, 32, 0, stream>>>(partials, out + (size_t)BB * NN * DD);
}
